// GCN_15633680957564
// MI455X (gfx1250) — matrix operand order repaired and checked by reading, not compiled, not run on hardware
//
#include <hip/hip_runtime.h>
#include <hip/hip_bf16.h>

// ---------------------------------------------------------------------------
// CDNA5 (gfx1250) 2-layer GCN.
//   Layer1: aggregate (128-wide) -> WMMA bf16 GEMM 128->512 + bias + ReLU
//   Layer2: WMMA bf16 GEMM 512->128 -> aggregate -> bias + ReLU
// Aggregation uses native GLOBAL_ATOMIC_ADD_F32; GEMMs use
// v_wmma_f32_16x16x32_bf16 (wave32). Each wave computes a 16x64 C strip
// (4 N-tiles): per K-step it loads A once plus 4 B frags (all loads issued
// before the 4 WMMAs, so waits stay partial and loads overlap compute).
// ---------------------------------------------------------------------------

typedef __attribute__((ext_vector_type(16))) __bf16 v16bf;
typedef __attribute__((ext_vector_type(8)))  float  v8f;

union FragU { unsigned int u[8]; v16bf v; };

__device__ __forceinline__ unsigned short f2bf_rne(float f) {
  unsigned int x = __builtin_bit_cast(unsigned int, f);
  x += 0x7fffu + ((x >> 16) & 1u);           // round-to-nearest-even
  return (unsigned short)(x >> 16);
}

__device__ __forceinline__ void atomAddF(float* p, float v) {
  __hip_atomic_fetch_add(p, v, __ATOMIC_RELAXED, __HIP_MEMORY_SCOPE_AGENT);
}

// --------------------------- utility kernels -------------------------------

__global__ void __launch_bounds__(256) k_zero_f32(float* __restrict__ p, long long n) {
  long long i = (long long)blockIdx.x * blockDim.x + threadIdx.x;
  long long s = (long long)gridDim.x * blockDim.x;
  for (; i < n; i += s) p[i] = 0.0f;
}

__global__ void __launch_bounds__(256) k_count_deg(const int* __restrict__ dst,
                                                   float* __restrict__ deg, int E) {
  int i = blockIdx.x * blockDim.x + threadIdx.x;
  if (i < E) atomAddF(&deg[dst[i]], 1.0f);
}

__global__ void __launch_bounds__(256) k_dinv(float* __restrict__ d, int n) {
  int i = blockIdx.x * blockDim.x + threadIdx.x;
  if (i < n) d[i] = rsqrtf(d[i] + 1.0f);     // +1 = self loop
}

__global__ void __launch_bounds__(256) k_f32_to_bf16(const float* __restrict__ in,
                                                     unsigned short* __restrict__ out,
                                                     long long n) {
  long long i = (long long)blockIdx.x * blockDim.x + threadIdx.x;
  long long s = (long long)gridDim.x * blockDim.x;
  for (; i < n; i += s) out[i] = f2bf_rne(in[i]);
}

// Scatter-add aggregation over E edges plus N self loops. One 128-thread block
// per item: thread f handles feature f (coalesced load + coalesced atomic).
__global__ void __launch_bounds__(128) k_aggregate(const int* __restrict__ src,
                                                   const int* __restrict__ dst,
                                                   const float* __restrict__ dinv,
                                                   const float* __restrict__ feat,
                                                   float* __restrict__ out,
                                                   int E, int N, int F) {
  long long e = blockIdx.x;
  if (e >= (long long)E + N) return;
  int s, d;
  if (e < E) { s = src[e]; d = dst[e]; }
  else       { s = d = (int)(e - E); }
  float w = dinv[s] * dinv[d];
  int f = threadIdx.x;
  atomAddF(&out[(long long)d * F + f], feat[(long long)s * F + f] * w);
}

__global__ void __launch_bounds__(256) k_bias_relu(float* __restrict__ out,
                                                   const float* __restrict__ b,
                                                   long long n, int F) {
  long long i = (long long)blockIdx.x * blockDim.x + threadIdx.x;
  long long s = (long long)gridDim.x * blockDim.x;
  for (; i < n; i += s) {
    float v = out[i] + b[(int)(i % F)];
    out[i] = v > 0.0f ? v : 0.0f;
  }
}

// --------------------------- WMMA GEMM -------------------------------------
// C[M x Nout] = A[M x K] * W[Nout x K]^T  (both bf16, K contiguous in memory)
// 256 threads = 8 waves per block; each wave owns a 16x64 strip of C
// (4 consecutive 16x16 N-tiles), loops K in steps of 32 with
// v_wmma_f32_16x16x32_bf16. Per K-step: issue ALL loads (A + 4xB) into
// distinct registers, then the 4 WMMAs, so waits are partial.
// BF16OUT: epilogue adds bias, applies ReLU, stores bf16. Else stores f32.
template <bool BF16OUT>
__global__ void __launch_bounds__(256)
k_gemm_bf16(const unsigned short* __restrict__ A,
            const unsigned short* __restrict__ Wt,
            const float* __restrict__ bias,
            void* __restrict__ Cout,
            int M, int K, int Nout) {
  const int lane = threadIdx.x & 31;
  const int wave = threadIdx.x >> 5;
  const int nGroups = Nout >> 6;             // 64-wide N groups
  const int totTasks = (M >> 4) * nGroups;
  const int t = blockIdx.x * 8 + wave;       // wave-uniform => EXEC all-1s
  if (t >= totTasks) return;
  const int mt = t / nGroups, ng = t - mt * nGroups;
  const int m0 = mt << 4, n0 = ng << 6;
  const int half = lane >> 4;                // lane group (0: lanes 0-15)
  const int idx  = lane & 15;

  v8f acc[4];
#pragma unroll
  for (int j = 0; j < 4; ++j) acc[j] = (v8f){0.f,0.f,0.f,0.f,0.f,0.f,0.f,0.f};

  // A fragment: lane holds row (m0+idx); VGPR p holds K pair
  //   K = 2p + 16*half
  const unsigned short* arow = A + (size_t)(m0 + idx) * K;
  // B fragment j: lane holds column (n0 + 16j + idx); element e -> K = 16*half + e
  const unsigned short* brow = Wt + (size_t)(n0 + idx) * K + half * 16;
  const size_t bstride = (size_t)16 * K;     // next 16x16 N-tile

  for (int k0 = 0; k0 < K; k0 += 32) {
    // ---- load phase: A frag + 4 B frags, all into live registers ----
    FragU fa;
    const unsigned int* a32 = (const unsigned int*)(arow + k0);  // 4B aligned
#pragma unroll
    for (int p = 0; p < 8; ++p) {
      int kpair = p + half * 8;
      fa.u[p] = a32[kpair];
    }
    FragU fb[4];
#pragma unroll
    for (int j = 0; j < 4; ++j) {
      const uint4* b128 = (const uint4*)(brow + j * bstride + k0); // 32B aligned
      uint4 q0 = b128[0], q1 = b128[1];
      fb[j].u[0] = q0.x; fb[j].u[1] = q0.y; fb[j].u[2] = q0.z; fb[j].u[3] = q0.w;
      fb[j].u[4] = q1.x; fb[j].u[5] = q1.y; fb[j].u[6] = q1.z; fb[j].u[7] = q1.w;
    }
    // ---- compute phase: 4 WMMAs sharing the A fragment ----
#pragma unroll
    for (int j = 0; j < 4; ++j) {
      acc[j] = __builtin_amdgcn_wmma_f32_16x16x32_bf16(
          /*neg_a=*/false, fa.v, /*neg_b=*/false, fb[j].v,
          /*c_mod=*/(short)0, acc[j], /*reuse_a=*/false, /*reuse_b=*/false);
    }
  }

  // C/D layout per tile: VGPR r -> M = m0 + r + 8*half, N = ntile0 + idx
#pragma unroll
  for (int j = 0; j < 4; ++j) {
    const int n = n0 + j * 16 + idx;
#pragma unroll
    for (int r = 0; r < 8; ++r) {
      const int m = m0 + r + 8 * half;
      float v = acc[j][r];
      if (BF16OUT) {
        v += bias[n];
        v = v > 0.0f ? v : 0.0f;
        ((unsigned short*)Cout)[(size_t)m * Nout + n] = f2bf_rne(v);
      } else {
        ((float*)Cout)[(size_t)m * Nout + n] = v;
      }
    }
  }
}

// --------------------------- host driver -----------------------------------

static inline size_t al256(size_t x) { return (x + 255) & ~(size_t)255; }

extern "C" void kernel_launch(void* const* d_in, const int* in_sizes, int n_in,
                              void* d_out, int out_size, void* d_ws, size_t ws_size,
                              hipStream_t stream) {
  (void)n_in; (void)ws_size;
  const float* x    = (const float*)d_in[0];
  const int*   eidx = (const int*)d_in[1];   // [2, E]; harness delivers int32
  const float* W1   = (const float*)d_in[2]; // [FHID, FIN]
  const float* b1   = (const float*)d_in[3];
  const float* W2   = (const float*)d_in[4]; // [FIN, FHID]
  const float* b2   = (const float*)d_in[5];
  float* out = (float*)d_out;

  const int FIN  = 128;
  const int N    = in_sizes[0] / FIN;        // 50000 (multiple of 16)
  const int E    = in_sizes[1] / 2;          // 600000
  const int FHID = in_sizes[3];              // 512
  const int* src = eidx;
  const int* dst = eidx + E;

  // workspace carve-up (~116 MB; fits MI455X 192 MB L2)
  char* ws = (char*)d_ws;
  size_t off = 0;
  float*          dinv = (float*)(ws + off);          off += al256((size_t)N * 4);
  float*          AX   = (float*)(ws + off);          off += al256((size_t)N * FIN * 4);
  unsigned short* AXbf = (unsigned short*)(ws + off); off += al256((size_t)N * FIN * 2);
  unsigned short* W1bf = (unsigned short*)(ws + off); off += al256((size_t)FHID * FIN * 2);
  unsigned short* W2bf = (unsigned short*)(ws + off); off += al256((size_t)FIN * FHID * 2);
  unsigned short* H1bf = (unsigned short*)(ws + off); off += al256((size_t)N * FHID * 2);
  float*          XW2  = (float*)(ws + off);          off += al256((size_t)N * FIN * 4);

  const long long nAX = (long long)N * FIN;
  auto blk = [](long long n, int t) { return (int)((n + t - 1) / t); };

  // 1) degrees -> dinv
  k_zero_f32<<<blk(N, 256), 256, 0, stream>>>(dinv, N);
  k_count_deg<<<blk(E, 256), 256, 0, stream>>>(dst, dinv, E);
  k_dinv<<<blk(N, 256), 256, 0, stream>>>(dinv, N);

  // 2) layer-1 aggregation first (narrow 128-wide features)
  k_zero_f32<<<blk(nAX, 256), 256, 0, stream>>>(AX, nAX);
  k_aggregate<<<E + N, 128, 0, stream>>>(src, dst, dinv, x, AX, E, N, FIN);

  // 3) bf16 conversions
  k_f32_to_bf16<<<blk(nAX, 256), 256, 0, stream>>>(AX, AXbf, nAX);
  k_f32_to_bf16<<<blk((long long)FHID * FIN, 256), 256, 0, stream>>>(W1, W1bf, (long long)FHID * FIN);
  k_f32_to_bf16<<<blk((long long)FIN * FHID, 256), 256, 0, stream>>>(W2, W2bf, (long long)FIN * FHID);

  // 4) GEMM1: H1 = relu(AX @ W1^T + b1), bf16 out   [N x FHID]
  {
    int tasks = (N >> 4) * (FHID >> 6);
    k_gemm_bf16<true><<<(tasks + 7) / 8, 256, 0, stream>>>(AXbf, W1bf, b1, H1bf, N, FIN, FHID);
  }

  // 5) GEMM2: XW2 = H1 @ W2^T, f32 out              [N x FIN]
  {
    int tasks = (N >> 4) * (FIN >> 6);
    k_gemm_bf16<false><<<(tasks + 7) / 8, 256, 0, stream>>>(H1bf, W2bf, nullptr, XW2, N, FHID, FIN);
  }

  // 6) layer-2 aggregation into output, then bias + ReLU
  k_zero_f32<<<blk(nAX, 256), 256, 0, stream>>>(out, nAX);
  k_aggregate<<<E + N, 128, 0, stream>>>(src, dst, dinv, XW2, out, E, N, FIN);
  k_bias_relu<<<blk(nAX, 256), 256, 0, stream>>>(out, b2, nAX, FIN);

  (void)out_size;
}
